// GPS_76158360092698
// MI455X (gfx1250) — compile-verified, matrix operand order checked by reading, not run on hardware
//
#include <hip/hip_runtime.h>
#include <hip/hip_bf16.h>

// ---------------------------------------------------------------------------
// Types for CDNA5 WMMA / TDM (gfx1250, wave32)
// ---------------------------------------------------------------------------
typedef __bf16 bf16_t;
typedef __attribute__((ext_vector_type(16))) __bf16 v16bf;
typedef __attribute__((ext_vector_type(8)))  __bf16 v8bf;
typedef __attribute__((ext_vector_type(8)))  float  v8f;
typedef unsigned int u32_t;
typedef u32_t __attribute__((ext_vector_type(4))) v4u;
typedef int   __attribute__((ext_vector_type(4))) v4i;
typedef int   __attribute__((ext_vector_type(8))) v8i;

#if defined(__has_builtin)
# if __has_builtin(__builtin_amdgcn_tensor_load_to_lds)
#  define ATH_HAVE_TDM 1
# endif
#endif

#define N_NODES 32768
#define C_DIM   128
#define E_EDGES 262144
#define B_GR    64
#define NPG     512
#define H_HEADS 4
#define DH      32
#define L_LAYERS 4
#define BN_EPS  1e-5f

// ---------------------------------------------------------------------------
// Small utility kernels
// ---------------------------------------------------------------------------
__global__ void zero_f32_kernel(float* __restrict__ p, int n) {
    int i = blockIdx.x * blockDim.x + threadIdx.x;
    if (i < n) p[i] = 0.0f;
}

__global__ void copy_f32_kernel(const float* __restrict__ a, float* __restrict__ b, int n4) {
    int i = blockIdx.x * blockDim.x + threadIdx.x;
    if (i < n4) ((float4*)b)[i] = ((const float4*)a)[i];
}

__global__ void add_f32_kernel(const float* __restrict__ a, const float* __restrict__ b,
                               float* __restrict__ o, int n4) {
    int i = blockIdx.x * blockDim.x + threadIdx.x;
    if (i < n4) {
        float4 x = ((const float4*)a)[i];
        float4 y = ((const float4*)b)[i];
        float4 z = make_float4(x.x + y.x, x.y + y.y, x.z + y.z, x.w + y.w);
        ((float4*)o)[i] = z;
    }
}

// transpose + convert weight: src f32 [K, Nc] row-major  ->  dst bf16 [Nc, K]
__global__ void prep_wt_kernel(const float* __restrict__ w, bf16_t* __restrict__ wt,
                               int K, int Nc) {
    int idx = blockIdx.x * blockDim.x + threadIdx.x;
    if (idx >= K * Nc) return;
    int n = idx / K;
    int k = idx - n * K;
    wt[idx] = (bf16_t)w[(size_t)k * Nc + n];
}

// ---------------------------------------------------------------------------
// PE BatchNorm stats (WALK=20 features over N rows) -> stat[0:20]=mean, [20:40]=rstd
// ---------------------------------------------------------------------------
__global__ void pe_stats_kernel(const float* __restrict__ pe, float* __restrict__ stat) {
    __shared__ float sh[256], sh2[256];
    int w = blockIdx.x;
    float s = 0.f, s2 = 0.f;
    for (int i = threadIdx.x; i < N_NODES; i += 256) {
        float v = pe[(size_t)i * 20 + w];
        s += v; s2 += v * v;
    }
    sh[threadIdx.x] = s; sh2[threadIdx.x] = s2;
    __syncthreads();
    for (int off = 128; off > 0; off >>= 1) {
        if (threadIdx.x < off) {
            sh[threadIdx.x]  += sh[threadIdx.x + off];
            sh2[threadIdx.x] += sh2[threadIdx.x + off];
        }
        __syncthreads();
    }
    if (threadIdx.x == 0) {
        float m = sh[0] * (1.0f / N_NODES);
        float var = sh2[0] * (1.0f / N_NODES) - m * m;
        stat[w] = m;
        stat[20 + w] = rsqrtf(var + BN_EPS);
    }
}

// h = concat(node_emb[x], pe_norm @ pe_lin_w + pe_lin_b)
__global__ void build_h_kernel(const int* __restrict__ x, const float* __restrict__ pe,
                               const float* __restrict__ stat,
                               const float* __restrict__ node_emb,
                               const float* __restrict__ plw, const float* __restrict__ plb,
                               const float* __restrict__ png, const float* __restrict__ pnb,
                               float* __restrict__ h) {
    int idx = blockIdx.x * blockDim.x + threadIdx.x;
    if (idx >= N_NODES * C_DIM) return;
    int i = idx >> 7;
    int c = idx & 127;
    float v;
    if (c < 120) {
        v = node_emb[x[i] * 120 + c];
    } else {
        int o = c - 120;
        float acc = plb[o];
        #pragma unroll
        for (int w = 0; w < 20; ++w) {
            float pn = (pe[(size_t)i * 20 + w] - stat[w]) * stat[20 + w] * png[w] + pnb[w];
            acc += pn * plw[w * 8 + o];
        }
        v = acc;
    }
    h[idx] = v;
}

// ---------------------------------------------------------------------------
// Edge message: agg[dst] += relu(h[src] + edge_emb[attr]);  agg pre-initialized to h
// ---------------------------------------------------------------------------
__global__ __launch_bounds__(256)
void edge_msg_kernel(const int* __restrict__ ei, const int* __restrict__ eattr,
                     const float* __restrict__ edge_emb,
                     const float* __restrict__ h, float* __restrict__ agg) {
    int tid = blockIdx.x * blockDim.x + threadIdx.x;
    int e  = tid >> 5;
    int c4 = (tid & 31) * 4;
    if (e >= E_EDGES) return;
    int src = ei[e];
    int dst = ei[E_EDGES + e];
    int at  = eattr[e];
    float4 hv = *(const float4*)(h + (size_t)src * C_DIM + c4);
    float4 ev = *(const float4*)(edge_emb + (size_t)at * C_DIM + c4);
    float m0 = fmaxf(hv.x + ev.x, 0.f);
    float m1 = fmaxf(hv.y + ev.y, 0.f);
    float m2 = fmaxf(hv.z + ev.z, 0.f);
    float m3 = fmaxf(hv.w + ev.w, 0.f);
    float* d = agg + (size_t)dst * C_DIM + c4;
    atomicAdd(d + 0, m0);
    atomicAdd(d + 1, m1);
    atomicAdd(d + 2, m2);
    atomicAdd(d + 3, m3);
}

// ---------------------------------------------------------------------------
// BatchNorm over node axis (C=128 channels): stats + apply
// ---------------------------------------------------------------------------
__global__ __launch_bounds__(128)
void col_stats_kernel(const float* __restrict__ t, float* __restrict__ sums) {
    int c = threadIdx.x;
    int r0 = blockIdx.x * 128;
    float s = 0.f, s2 = 0.f;
    #pragma unroll 4
    for (int r = 0; r < 128; ++r) {
        float v = t[(size_t)(r0 + r) * C_DIM + c];
        s += v; s2 += v * v;
    }
    atomicAdd(&sums[c], s);
    atomicAdd(&sums[C_DIM + c], s2);
}

__global__ void bn_apply_kernel(const float* __restrict__ t, const float* __restrict__ sums,
                                const float* __restrict__ g, const float* __restrict__ b,
                                float* __restrict__ out) {
    int idx = blockIdx.x * blockDim.x + threadIdx.x;
    if (idx >= N_NODES * C_DIM) return;
    int c = idx & 127;
    float mean = sums[c] * (1.0f / N_NODES);
    float var  = sums[C_DIM + c] * (1.0f / N_NODES) - mean * mean;
    out[idx] = (t[idx] - mean) * rsqrtf(var + BN_EPS) * g[c] + b[c];
}

// ---------------------------------------------------------------------------
// Generic bf16 WMMA GEMM:  out[M,Nc] = op(A[M,K] @ W[K,Nc] + bias (+residual))
// A: f32 streamed from global, converted in-register to bf16.
// Wt: bf16 pre-transposed [Nc,K]; per-block 64xK tile staged into LDS by the
// Tensor Data Mover (D# built per CDNA5 ISA ch.8) with a 16B/row pad skew so
// the 16-lane x 16B B-operand fetches are bank-conflict free.
// 1 wave = 16x64 output tile (4 accumulators); block = 4 waves; grid (M/64, Nc/64)
// ---------------------------------------------------------------------------
__global__ __launch_bounds__(128)
void gemm_bf16_kernel(const float* __restrict__ A, const bf16_t* __restrict__ Wt,
                      const float* __restrict__ bias, const float* __restrict__ residual,
                      float* __restrict__ out, int M, int K, int Nc, int relu) {
    __shared__ __align__(16) bf16_t lds_w[64 * 264];   // 64 rows, K<=256, +8 el pad/row
    const int lane = threadIdx.x & 31;
    const int wave = threadIdx.x >> 5;
    const int mt   = blockIdx.x * 4 + wave;   // 16-row tile index
    const int n0   = blockIdx.y * 64;
    const int row  = lane & 15;
    const int kh   = lane >> 4;               // which K-half this lane holds
    const int ldk  = K + 8;                   // padded LDS row stride (elements)

    // ---- stage weight tile Wt[n0 .. n0+63][0..K) into LDS ----
#if defined(ATH_HAVE_TDM)
    if (threadIdx.x < 32) {   // one wave issues the TDM op (EXEC ignored by TDM)
        unsigned long long ga = (unsigned long long)(uintptr_t)(Wt + (size_t)n0 * K);
        u32_t lds_off = (u32_t)(uintptr_t)(&lds_w[0]);
        u32_t nelem = (u32_t)(64 * K);                       // contiguous bf16 elements
        u32_t icode = (K == 128) ? 5u : 6u;                  // pad every K/2 DWORDs
        v4u g0; v8i g1; v4i g2; v4i g3;
        g0[0] = 1u;                                          // count=1 (valid user D#)
        g0[1] = lds_off;                                     // lds_addr (bytes)
        g0[2] = (u32_t)(ga & 0xFFFFFFFFu);                   // global_addr[31:0]
        g0[3] = (u32_t)((ga >> 32) & 0x1FFFFFFu) | (2u << 30); // global_addr[56:32] | type=2
        g1[0] = (int)((1u << 16) | (1u << 20) | (icode << 22) | (3u << 25));
                 // data_size=2B | pad_enable | pad_interval | pad_amount=4 DWORDs (16B)
        g1[1] = (int)((nelem & 0xFFFFu) << 16);              // tensor_dim0[15:0]
        g1[2] = (int)((nelem >> 16) | (1u << 16));           // tensor_dim0[31:16] | tensor_dim1=1
        g1[3] = (int)((nelem & 0xFFFFu) << 16);              // tile_dim0 = nelem (<=16384)
        g1[4] = 1;                                           // tile_dim1=1, tile_dim2=0
        g1[5] = (int)nelem;                                  // tensor_dim0_stride
        g1[6] = 0;
        g1[7] = 0;
        g2[0] = g2[1] = g2[2] = g2[3] = 0;
        g3[0] = g3[1] = g3[2] = g3[3] = 0;
#if __clang_major__ >= 23
        v8i g4;
        #pragma unroll
        for (int t = 0; t < 8; ++t) g4[t] = 0;
        __builtin_amdgcn_tensor_load_to_lds(g0, g1, g2, g3, g4, 0);
#else
        __builtin_amdgcn_tensor_load_to_lds(g0, g1, g2, g3, 0);
#endif
        __builtin_amdgcn_s_wait_tensorcnt(0);
    }
#else
    // fallback: cooperative copy with the same padded layout
    for (int i = threadIdx.x; i < 64 * (K >> 3); i += 128) {
        int r = i / (K >> 3), seg = i - r * (K >> 3);
        *(v8bf*)(lds_w + (size_t)r * ldk + seg * 8) =
            *(const v8bf*)(Wt + (size_t)(n0 + r) * K + seg * 8);
    }
#endif
    __syncthreads();

    v8f acc[4];
    #pragma unroll
    for (int j = 0; j < 4; ++j)
        #pragma unroll
        for (int r = 0; r < 8; ++r) acc[j][r] = 0.0f;

    const float* arow = A + (size_t)(mt * 16 + row) * K;
    for (int kk = 0; kk < K; kk += 32) {
        // A tile (16x32): per-lane 16 bf16, K pattern {kh*8.., 16+kh*8..}
        v16bf a;
        const float* ap0 = arow + kk + kh * 8;
        const float* ap1 = arow + kk + 16 + kh * 8;
        #pragma unroll
        for (int t = 0; t < 8; ++t) {
            a[t]     = (bf16_t)ap0[t];
            a[8 + t] = (bf16_t)ap1[t];
        }
        #pragma unroll
        for (int j = 0; j < 4; ++j) {
            const bf16_t* wr = lds_w + (size_t)(j * 16 + row) * ldk + kk;  // ds_load_b128
            v16bf bmat;
            *(v8bf*)&bmat       = *(const v8bf*)(wr + kh * 8);
            *((v8bf*)&bmat + 1) = *(const v8bf*)(wr + 16 + kh * 8);
            acc[j] = __builtin_amdgcn_wmma_f32_16x16x32_bf16(
                false, a, false, bmat, (short)0, acc[j], false, false);
        }
    }
    // epilogue: C layout -> VGPR r holds row (r + kh*8), col = lane&15
    #pragma unroll
    for (int j = 0; j < 4; ++j) {
        int col = n0 + j * 16 + row;
        float bv = bias ? bias[col] : 0.0f;
        #pragma unroll
        for (int r = 0; r < 8; ++r) {
            int mr = mt * 16 + r + kh * 8;
            size_t idx = (size_t)mr * Nc + col;
            float v = acc[j][r] + bv;
            if (residual) v += residual[idx];
            if (relu) v = fmaxf(v, 0.0f);
            out[idx] = v;
        }
    }
}

// ---------------------------------------------------------------------------
// V transpose for attention: Vt[b][h][d][j] (bf16) from qkv f32 [N, 384]
// ---------------------------------------------------------------------------
__global__ void prep_vt_kernel(const float* __restrict__ qkv, bf16_t* __restrict__ vt) {
    int idx = blockIdx.x * blockDim.x + threadIdx.x;
    if (idx >= B_GR * H_HEADS * DH * NPG) return;
    int j  = idx & 511;
    int d  = (idx >> 9) & 31;
    int hd = (idx >> 14) & 3;
    int b  = idx >> 16;
    vt[idx] = (bf16_t)qkv[((size_t)(b * NPG + j)) * 384 + 256 + hd * DH + d];
}

// ---------------------------------------------------------------------------
// Flash attention: 1 wave = 16 query rows of one (b,h); streams keys in 32-chunks.
// S = Q*K^T via 2 WMMA, online softmax with half-wave shfl reductions,
// P re-layout (C-layout -> A-layout) through LDS, O += P*V via 2 WMMA.
// grid = (B*H, NPG/64), block = 128 (4 waves)
// ---------------------------------------------------------------------------
__global__ __launch_bounds__(128)
void attn_kernel(const float* __restrict__ qkv, const bf16_t* __restrict__ Vt,
                 float* __restrict__ o_out) {
    __shared__ __align__(16) bf16_t Pl[4][16][32];
    const int lane = threadIdx.x & 31;
    const int wave = threadIdx.x >> 5;
    const int bh = blockIdx.x;
    const int b  = bh >> 2;
    const int hd = bh & 3;
    const int m0 = blockIdx.y * 64 + wave * 16;
    const int row = lane & 15;
    const int kh  = lane >> 4;
    const float scale = 0.17677669529663687f;   // 1/sqrt(32)

    const float* qrow = qkv + ((size_t)(b * NPG + m0 + row)) * 384 + hd * DH;
    v16bf aq;
    #pragma unroll
    for (int t = 0; t < 8; ++t) {
        aq[t]     = (bf16_t)qrow[kh * 8 + t];
        aq[8 + t] = (bf16_t)qrow[16 + kh * 8 + t];
    }

    v8f o0, o1;
    float mrow[8], lrow[8];
    #pragma unroll
    for (int r = 0; r < 8; ++r) { o0[r] = 0.f; o1[r] = 0.f; mrow[r] = -1e30f; lrow[r] = 0.f; }

    const bf16_t* vbase = Vt + (size_t)bh * DH * NPG;

    for (int j0 = 0; j0 < NPG; j0 += 32) {
        v8f s0, s1;
        #pragma unroll
        for (int r = 0; r < 8; ++r) { s0[r] = 0.f; s1[r] = 0.f; }
        {
            const float* kr = qkv + ((size_t)(b * NPG + j0 + row)) * 384 + 128 + hd * DH;
            v16bf bk;
            #pragma unroll
            for (int t = 0; t < 8; ++t) {
                bk[t]     = (bf16_t)kr[kh * 8 + t];
                bk[8 + t] = (bf16_t)kr[16 + kh * 8 + t];
            }
            s0 = __builtin_amdgcn_wmma_f32_16x16x32_bf16(false, aq, false, bk, (short)0, s0, false, false);
        }
        {
            const float* kr = qkv + ((size_t)(b * NPG + j0 + 16 + row)) * 384 + 128 + hd * DH;
            v16bf bk;
            #pragma unroll
            for (int t = 0; t < 8; ++t) {
                bk[t]     = (bf16_t)kr[kh * 8 + t];
                bk[8 + t] = (bf16_t)kr[16 + kh * 8 + t];
            }
            s1 = __builtin_amdgcn_wmma_f32_16x16x32_bf16(false, aq, false, bk, (short)0, s1, false, false);
        }
        #pragma unroll
        for (int r = 0; r < 8; ++r) {
            float v0 = s0[r] * scale;
            float v1 = s1[r] * scale;
            float mx = fmaxf(v0, v1);
            mx = fmaxf(mx, __shfl_xor(mx, 1, 32));
            mx = fmaxf(mx, __shfl_xor(mx, 2, 32));
            mx = fmaxf(mx, __shfl_xor(mx, 4, 32));
            mx = fmaxf(mx, __shfl_xor(mx, 8, 32));
            float mnew = fmaxf(mrow[r], mx);
            float corr = __expf(mrow[r] - mnew);
            float p0 = __expf(v0 - mnew);
            float p1 = __expf(v1 - mnew);
            float ps = p0 + p1;
            ps += __shfl_xor(ps, 1, 32);
            ps += __shfl_xor(ps, 2, 32);
            ps += __shfl_xor(ps, 4, 32);
            ps += __shfl_xor(ps, 8, 32);
            lrow[r] = lrow[r] * corr + ps;
            mrow[r] = mnew;
            o0[r] *= corr;
            o1[r] *= corr;
            Pl[wave][r + kh * 8][row]      = (bf16_t)p0;
            Pl[wave][r + kh * 8][16 + row] = (bf16_t)p1;
        }
        // wave-local LDS RAW hazard: split CDNA5 DS counter
        asm volatile("s_wait_dscnt 0x0" ::: "memory");
        v16bf ap;
        *(v8bf*)&ap       = *(const v8bf*)&Pl[wave][row][kh * 8];
        *((v8bf*)&ap + 1) = *(const v8bf*)&Pl[wave][row][16 + kh * 8];
        v16bf bv0, bv1;
        {
            const bf16_t* vp = vbase + (size_t)row * NPG + j0;
            *(v8bf*)&bv0       = *(const v8bf*)(vp + kh * 8);
            *((v8bf*)&bv0 + 1) = *(const v8bf*)(vp + 16 + kh * 8);
        }
        {
            const bf16_t* vp = vbase + (size_t)(16 + row) * NPG + j0;
            *(v8bf*)&bv1       = *(const v8bf*)(vp + kh * 8);
            *((v8bf*)&bv1 + 1) = *(const v8bf*)(vp + 16 + kh * 8);
        }
        o0 = __builtin_amdgcn_wmma_f32_16x16x32_bf16(false, ap, false, bv0, (short)0, o0, false, false);
        o1 = __builtin_amdgcn_wmma_f32_16x16x32_bf16(false, ap, false, bv1, (short)0, o1, false, false);
    }
    #pragma unroll
    for (int r = 0; r < 8; ++r) {
        int i = m0 + r + kh * 8;
        float inv = 1.0f / lrow[r];
        float* dst = o_out + ((size_t)(b * NPG + i)) * C_DIM + hd * DH;
        dst[row]      = o0[r] * inv;
        dst[16 + row] = o1[r] * inv;
    }
}

// ---------------------------------------------------------------------------
// Pool (global_add_pool) and final MLP
// ---------------------------------------------------------------------------
__global__ __launch_bounds__(128)
void pool_kernel(const float* __restrict__ h, float* __restrict__ g) {
    int b = blockIdx.x, c = threadIdx.x;
    float s = 0.f;
    for (int i = 0; i < NPG; ++i) s += h[((size_t)(b * NPG + i)) * C_DIM + c];
    g[b * C_DIM + c] = s;
}

__global__ void mlp_kernel(const float* __restrict__ g,
                           const float* __restrict__ w1, const float* __restrict__ b1,
                           const float* __restrict__ w2, const float* __restrict__ b2,
                           const float* __restrict__ w3, const float* __restrict__ b3,
                           float* __restrict__ out) {
    int b = threadIdx.x;
    if (b >= B_GR) return;
    float y1[64];
    for (int o = 0; o < 64; ++o) {
        float a = b1[o];
        for (int k = 0; k < 128; ++k) a += g[b * 128 + k] * w1[k * 64 + o];
        y1[o] = fmaxf(a, 0.f);
    }
    float y2[32];
    for (int o = 0; o < 32; ++o) {
        float a = b2[o];
        for (int k = 0; k < 64; ++k) a += y1[k] * w2[k * 32 + o];
        y2[o] = fmaxf(a, 0.f);
    }
    float a = b3[0];
    for (int k = 0; k < 32; ++k) a += y2[k] * w3[k];
    out[b] = a;
}

// ---------------------------------------------------------------------------
// Host orchestration
// ---------------------------------------------------------------------------
extern "C" void kernel_launch(void* const* d_in, const int* in_sizes, int n_in,
                              void* d_out, int out_size, void* d_ws, size_t ws_size,
                              hipStream_t stream) {
    (void)in_sizes; (void)n_in; (void)out_size; (void)ws_size;
    const int*   x         = (const int*)  d_in[0];
    const float* pe        = (const float*)d_in[1];
    const int*   eidx      = (const int*)  d_in[2];
    const int*   eattr     = (const int*)  d_in[3];
    /* batch = d_in[4] : implicit (contiguous equal-size graphs) */
    const float* node_emb  = (const float*)d_in[5];
    const float* pe_lin_w  = (const float*)d_in[6];
    const float* pe_lin_b  = (const float*)d_in[7];
    const float* pe_norm_g = (const float*)d_in[8];
    const float* pe_norm_b = (const float*)d_in[9];
    const float* edge_emb  = (const float*)d_in[10];
    const float* gine_w1   = (const float*)d_in[11];
    const float* gine_b1   = (const float*)d_in[12];
    const float* gine_w2   = (const float*)d_in[13];
    const float* gine_b2   = (const float*)d_in[14];
    const float* attn_wqkv = (const float*)d_in[15];
    const float* attn_bqkv = (const float*)d_in[16];
    const float* attn_wo   = (const float*)d_in[17];
    const float* attn_bo   = (const float*)d_in[18];
    const float* norm1_g   = (const float*)d_in[19];
    const float* norm1_b   = (const float*)d_in[20];
    const float* norm2_g   = (const float*)d_in[21];
    const float* norm2_b   = (const float*)d_in[22];
    const float* norm3_g   = (const float*)d_in[23];
    const float* norm3_b   = (const float*)d_in[24];
    const float* ffn_w1    = (const float*)d_in[25];
    const float* ffn_b1    = (const float*)d_in[26];
    const float* ffn_w2    = (const float*)d_in[27];
    const float* ffn_b2    = (const float*)d_in[28];
    const float* mlp_w1    = (const float*)d_in[29];
    const float* mlp_b1    = (const float*)d_in[30];
    const float* mlp_w2    = (const float*)d_in[31];
    const float* mlp_b2    = (const float*)d_in[32];
    const float* mlp_w3    = (const float*)d_in[33];
    const float* mlp_b3    = (const float*)d_in[34];

    char* base = (char*)d_ws;
    size_t off = 0;
    auto alloc = [&](size_t bytes) -> void* {
        off = (off + 255) & ~(size_t)255;
        void* p = base + off;
        off += bytes;
        return p;
    };
    float* stat_pe = (float*)alloc(40 * sizeof(float));
    float* bn_sums = (float*)alloc(2 * C_DIM * sizeof(float));
    bf16_t* g1t  = (bf16_t*)alloc((size_t)L_LAYERS * 128 * 128 * sizeof(bf16_t));
    bf16_t* g2t  = (bf16_t*)alloc((size_t)L_LAYERS * 128 * 128 * sizeof(bf16_t));
    bf16_t* qkt  = (bf16_t*)alloc((size_t)L_LAYERS * 384 * 128 * sizeof(bf16_t));
    bf16_t* wot  = (bf16_t*)alloc((size_t)L_LAYERS * 128 * 128 * sizeof(bf16_t));
    bf16_t* f1t  = (bf16_t*)alloc((size_t)L_LAYERS * 256 * 128 * sizeof(bf16_t));
    bf16_t* f2t  = (bf16_t*)alloc((size_t)L_LAYERS * 128 * 256 * sizeof(bf16_t));
    float* h    = (float*)alloc((size_t)N_NODES * 128 * sizeof(float));
    float* zbuf = (float*)alloc((size_t)N_NODES * 128 * sizeof(float));
    float* hloc = (float*)alloc((size_t)N_NODES * 128 * sizeof(float));
    float* buf4 = (float*)alloc((size_t)N_NODES * 128 * sizeof(float));
    float* bufA = (float*)alloc((size_t)N_NODES * 384 * sizeof(float));   // qkv / t1 / mid
    bf16_t* vt  = (bf16_t*)alloc((size_t)B_GR * H_HEADS * DH * NPG * sizeof(bf16_t));
    float* gpool = (float*)alloc((size_t)B_GR * C_DIM * sizeof(float));

    const int NC = N_NODES * C_DIM;
    const int NC4 = NC / 4;

    auto prep = [&](const float* src, bf16_t* dst, int K, int Nc) {
        int total = K * Nc;
        hipLaunchKernelGGL(prep_wt_kernel, dim3((total + 255) / 256), dim3(256), 0, stream,
                           src, dst, K, Nc);
    };
    for (int l = 0; l < L_LAYERS; ++l) {
        prep(gine_w1   + (size_t)l * 128 * 128, g1t + (size_t)l * 128 * 128, 128, 128);
        prep(gine_w2   + (size_t)l * 128 * 128, g2t + (size_t)l * 128 * 128, 128, 128);
        prep(attn_wqkv + (size_t)l * 128 * 384, qkt + (size_t)l * 384 * 128, 128, 384);
        prep(attn_wo   + (size_t)l * 128 * 128, wot + (size_t)l * 128 * 128, 128, 128);
        prep(ffn_w1    + (size_t)l * 128 * 256, f1t + (size_t)l * 256 * 128, 128, 256);
        prep(ffn_w2    + (size_t)l * 256 * 128, f2t + (size_t)l * 128 * 256, 256, 128);
    }

    hipLaunchKernelGGL(pe_stats_kernel, dim3(20), dim3(256), 0, stream, pe, stat_pe);
    hipLaunchKernelGGL(build_h_kernel, dim3((NC + 255) / 256), dim3(256), 0, stream,
                       x, pe, stat_pe, node_emb, pe_lin_w, pe_lin_b, pe_norm_g, pe_norm_b, h);

    auto gemm = [&](const float* A, const bf16_t* Wt, const float* bias,
                    const float* residual, float* out, int K, int Nc, int relu) {
        hipLaunchKernelGGL(gemm_bf16_kernel, dim3(N_NODES / 64, Nc / 64), dim3(128), 0, stream,
                           A, Wt, bias, residual, out, N_NODES, K, Nc, relu);
    };
    auto bn = [&](const float* t, const float* g, const float* b, float* out) {
        hipLaunchKernelGGL(zero_f32_kernel, dim3(1), dim3(256), 0, stream, bn_sums, 256);
        hipLaunchKernelGGL(col_stats_kernel, dim3(N_NODES / 128), dim3(128), 0, stream, t, bn_sums);
        hipLaunchKernelGGL(bn_apply_kernel, dim3((NC + 255) / 256), dim3(256), 0, stream,
                           t, bn_sums, g, b, out);
    };

    for (int l = 0; l < L_LAYERS; ++l) {
        // ---- GINE local branch ----
        hipLaunchKernelGGL(copy_f32_kernel, dim3((NC4 + 255) / 256), dim3(256), 0, stream,
                           h, zbuf, NC4);                         // z = h (folds the +h)
        hipLaunchKernelGGL(edge_msg_kernel, dim3(E_EDGES * 32 / 256), dim3(256), 0, stream,
                           eidx, eattr, edge_emb, h, zbuf);       // z += sum relu(h[src]+ea)
        gemm(zbuf, g1t + (size_t)l * 128 * 128, gine_b1 + l * 128, nullptr, hloc, 128, 128, 1);
        gemm(hloc, g2t + (size_t)l * 128 * 128, gine_b2 + l * 128, h, zbuf, 128, 128, 0);
        bn(zbuf, norm1_g + l * 128, norm1_b + l * 128, hloc);     // hloc = BN(local + h)

        // ---- global attention branch ----
        gemm(h, qkt + (size_t)l * 384 * 128, attn_bqkv + l * 384, nullptr, bufA, 128, 384, 0);
        hipLaunchKernelGGL(prep_vt_kernel, dim3(B_GR * H_HEADS * DH * NPG / 256), dim3(256),
                           0, stream, bufA, vt);
        hipLaunchKernelGGL(attn_kernel, dim3(B_GR * H_HEADS, NPG / 64), dim3(128), 0, stream,
                           bufA, vt, zbuf);                       // zbuf = attention output
        gemm(zbuf, wot + (size_t)l * 128 * 128, attn_bo + l * 128, h, buf4, 128, 128, 0);
        bn(buf4, norm2_g + l * 128, norm2_b + l * 128, zbuf);     // zbuf = BN(o + h)

        // ---- combine + FFN ----
        hipLaunchKernelGGL(add_f32_kernel, dim3((NC4 + 255) / 256), dim3(256), 0, stream,
                           hloc, zbuf, buf4, NC4);                // out = h_loc + h_att
        gemm(buf4, f1t + (size_t)l * 256 * 128, ffn_b1 + l * 256, nullptr, bufA, 128, 256, 1);
        gemm(bufA, f2t + (size_t)l * 128 * 256, ffn_b2 + l * 128, buf4, hloc, 256, 128, 0);
        bn(hloc, norm3_g + l * 128, norm3_b + l * 128, h);        // h = BN(out + ffn)
    }

    hipLaunchKernelGGL(pool_kernel, dim3(B_GR), dim3(128), 0, stream, h, gpool);
    hipLaunchKernelGGL(mlp_kernel, dim3(1), dim3(64), 0, stream,
                       gpool, mlp_w1, mlp_b1, mlp_w2, mlp_b2, mlp_w3, mlp_b3, (float*)d_out);
}
